// SelfAttention_55954833932670
// MI455X (gfx1250) — matrix operand order repaired and checked by reading, not compiled, not run on hardware
//
#include <hip/hip_runtime.h>

#define DEV __device__ __forceinline__

#define BATCH 2
#define SEQ   2048
#define DIMC  1024
#define NHEAD 16
#define HDIM  64
#define SCALE 0.125f   // 1/sqrt(64)
#define SCALE_LOG2E 0.1803368801111204f   // SCALE * log2(e)

typedef __attribute__((ext_vector_type(16))) __bf16        v16bf;
typedef __attribute__((ext_vector_type(8)))  float          v8f;
typedef __attribute__((ext_vector_type(8)))  unsigned short us8;
typedef __attribute__((ext_vector_type(16))) unsigned short us16;

union BF16x16 { us16 u; v16bf b; };

DEV float fast_exp2(float x) { return __builtin_amdgcn_exp2f(x); }

DEV unsigned short f2bf(float f) {
  unsigned u = __float_as_uint(f);
  u += 0x7fffu + ((u >> 16) & 1u);     // round-to-nearest-even
  return (unsigned short)(u >> 16);
}

DEV v16bf load_frag(const unsigned short* p0, const unsigned short* p1) {
  us8 lo = *(const us8*)p0;
  us8 hi = *(const us8*)p1;
  BF16x16 r;
#pragma unroll
  for (int i = 0; i < 8; ++i) { r.u[i] = lo[i]; r.u[i + 8] = hi[i]; }
  return r.b;
}

// A-fragment 16x32 bf16: lane row = lane&15; elems e<8 -> k=e+8g, e>=8 -> k=16+(e-8)+8g
DEV v16bf load_a(const unsigned short* row, int g) {
  return load_frag(row + 8 * g, row + 16 + 8 * g);
}
// B-fragment 32x16 bf16 read from row-major [N][K]: lane row = col n; elems e<8 -> k=e+8g, e>=8 -> k=16+(e-8)+8g
DEV v16bf load_b(const unsigned short* row, int g) {
  return load_frag(row + 8 * g, row + 16 + 8 * g);
}

DEV v8f wmma_bf16(v16bf a, v16bf b, v8f c) {
  return __builtin_amdgcn_wmma_f32_16x16x32_bf16(false, a, false, b, (short)0, c, false, false);
}

// async global -> LDS copy of 16 bytes per lane (gfx1250 ASYNCcnt path)
DEV void async_copy_b128(void* lds_dst, const void* gsrc) {
  unsigned ldsoff = (unsigned)(uintptr_t)lds_dst;   // low 32 bits of shared aperture addr = LDS offset
  asm volatile("global_load_async_to_lds_b128 %0, %1, off"
               :: "v"(ldsoff), "v"(gsrc) : "memory");
}
DEV void wait_async() { asm volatile("s_wait_asynccnt 0x0" ::: "memory"); }

// ---------------- conversion kernels ----------------
__global__ void convert_bf16_kernel(const float* __restrict__ in,
                                    unsigned short* __restrict__ out, int n) {
  int i = blockIdx.x * blockDim.x + threadIdx.x;
  if (i < n) out[i] = f2bf(in[i]);
}

// in: [R][C] f32 -> out: [C][R] bf16
__global__ void transpose_bf16_kernel(const float* __restrict__ in,
                                      unsigned short* __restrict__ out, int R, int C) {
  int i = blockIdx.x * blockDim.x + threadIdx.x;
  if (i < R * C) {
    int c = i / R, r = i % R;
    out[i] = f2bf(in[(size_t)r * C + c]);
  }
}

// ---------------- QKV GEMM: [B*L,DIM] x [DIM,3DIM] + bias -> Q,K,Vt (bf16) ----------------
// Block: 8 waves; each wave 16(M)x64(N); all waves share one 64-wide N stripe.
// B tile (64x32 bf16) staged in LDS via async copies, double buffered.
__global__ void __launch_bounds__(256) qkv_gemm_kernel(
    const unsigned short* __restrict__ xbf,   // [B*L][DIM]
    const unsigned short* __restrict__ wT,    // [3*DIM][DIM]
    const float* __restrict__ bias,           // [3*DIM]
    unsigned short* __restrict__ Q,           // [B][H][L][HDIM]
    unsigned short* __restrict__ K,           // [B][H][L][HDIM]
    unsigned short* __restrict__ Vt) {        // [B][H][HDIM][L]
  __shared__ unsigned short bTile[2][64][32];
  const int wave = threadIdx.x >> 5;
  const int lane = threadIdx.x & 31;
  const int g = lane >> 4, col = lane & 15;
  const int nblk = blockIdx.x % 48;
  const int mblk = blockIdx.x / 48;
  const int m0 = mblk * 128 + wave * 16;
  const int n0 = nblk * 64;

  // async-copy mapping: 256 threads x 16B = 4KB tile
  const int crow = threadIdx.x >> 2;            // 0..63 (n within stripe)
  const int ccol = (threadIdx.x & 3) * 8;       // 0,8,16,24 (k ushorts)
  const unsigned short* gB = wT + (size_t)(n0 + crow) * DIMC + ccol;

  async_copy_b128(&bTile[0][crow][ccol], gB);
  wait_async();
  __syncthreads();

  v8f acc[4];
#pragma unroll
  for (int t = 0; t < 4; ++t)
#pragma unroll
    for (int r = 0; r < 8; ++r) acc[t][r] = 0.f;

  const unsigned short* arow = xbf + (size_t)(m0 + col) * DIMC;
  for (int kt = 0; kt < DIMC; kt += 32) {
    const int buf = (kt >> 5) & 1;
    if (kt + 32 < DIMC)
      async_copy_b128(&bTile[buf ^ 1][crow][ccol], gB + kt + 32);
    v16bf a = load_a(arow + kt, g);
    v16bf b0 = load_b(&bTile[buf][col][0], g);
    v16bf b1 = load_b(&bTile[buf][16 + col][0], g);
    v16bf b2 = load_b(&bTile[buf][32 + col][0], g);
    v16bf b3 = load_b(&bTile[buf][48 + col][0], g);
    acc[0] = wmma_bf16(a, b0, acc[0]);
    acc[1] = wmma_bf16(a, b1, acc[1]);
    acc[2] = wmma_bf16(a, b2, acc[2]);
    acc[3] = wmma_bf16(a, b3, acc[3]);
    wait_async();
    __syncthreads();
  }

  const int bb = m0 / SEQ;
  const int row0 = m0 % SEQ;
#pragma unroll
  for (int t = 0; t < 4; ++t) {
    const int n = n0 + t * 16 + col;
    const float bv = bias[n];
    const int sec = (n0 + t * 16) / DIMC;           // uniform: 0=q,1=k,2=v
    const int nin = n % DIMC;
    const int h = nin / HDIM;
    const int d = nin % HDIM;
    const size_t hb = (size_t)(bb * NHEAD + h);
    if (sec == 2) {
      us8 pk;
#pragma unroll
      for (int r = 0; r < 8; ++r) pk[r] = f2bf(acc[t][r] + bv);
      *(us8*)(Vt + (hb * HDIM + d) * SEQ + row0 + 8 * g) = pk;
    } else {
      unsigned short* dst = (sec == 0 ? Q : K) + (hb * SEQ + row0 + 8 * g) * HDIM + d;
#pragma unroll
      for (int r = 0; r < 8; ++r) dst[(size_t)r * HDIM] = f2bf(acc[t][r] + bv);
    }
  }
}

// ---------------- flash attention: block = 4 waves = 64 q rows of one (b,h) ----------------
// K block (32x64) and Vt block (64x32) staged in LDS via async copies, double buffered.
__global__ void __launch_bounds__(128) attn_kernel(
    const unsigned short* __restrict__ Q,
    const unsigned short* __restrict__ K,
    const unsigned short* __restrict__ Vt,
    unsigned short* __restrict__ attn) {      // [B][L][DIM]
  __shared__ unsigned short kTile[2][32][64];
  __shared__ unsigned short vTile[2][64][32];
  __shared__ unsigned short pLDS[4][16][32];
  const int wave = threadIdx.x >> 5;
  const int lane = threadIdx.x & 31;
  const int g = lane >> 4, col = lane & 15;
  const int qchunks = SEQ / 64;                     // 32
  const int bb = blockIdx.x / (NHEAD * qchunks);
  const int h = (blockIdx.x / qchunks) % NHEAD;
  const int q0 = (blockIdx.x % qchunks) * 64 + wave * 16;

  const size_t hb = (size_t)(bb * NHEAD + h);
  const unsigned short* qrow = Q + (hb * SEQ + q0 + col) * HDIM;
  const v16bf qa0 = load_a(qrow, g);
  const v16bf qa1 = load_a(qrow + 32, g);

  const unsigned short* kbase = K + hb * SEQ * HDIM;
  const unsigned short* vbase = Vt + hb * HDIM * SEQ;

  // prologue: stage j0 = 0 into buffer 0
  {
    const int tid = threadIdx.x;
#pragma unroll
    for (int rep = 0; rep < 2; ++rep) {
      const int chunk = tid + rep * 128;            // 0..255
      async_copy_b128((unsigned short*)kTile[0] + chunk * 8, kbase + chunk * 8);
      const int vr = chunk >> 2, vc = (chunk & 3) * 8;
      async_copy_b128(&vTile[0][vr][vc], vbase + (size_t)vr * SEQ + vc);
    }
  }
  wait_async();
  __syncthreads();

  float m_i[8], l_i[8];
  v8f o[4];
#pragma unroll
  for (int r = 0; r < 8; ++r) { m_i[r] = -1e30f; l_i[r] = 0.f; }
#pragma unroll
  for (int t = 0; t < 4; ++t)
#pragma unroll
    for (int r = 0; r < 8; ++r) o[t][r] = 0.f;

  for (int j0 = 0; j0 < SEQ; j0 += 32) {
    const int buf = (j0 >> 5) & 1;
    if (j0 + 32 < SEQ) {
      const int tid = threadIdx.x;
#pragma unroll
      for (int rep = 0; rep < 2; ++rep) {
        const int chunk = tid + rep * 128;
        async_copy_b128((unsigned short*)kTile[buf ^ 1] + chunk * 8,
                        kbase + (size_t)(j0 + 32) * HDIM + chunk * 8);
        const int vr = chunk >> 2, vc = (chunk & 3) * 8;
        async_copy_b128(&vTile[buf ^ 1][vr][vc],
                        vbase + (size_t)vr * SEQ + (j0 + 32) + vc);
      }
    }

    // S = Q K^T for 32 keys (K fragments from LDS, all loaded before WMMA chain)
    v16bf kb00 = load_b(&kTile[buf][col][0], g);
    v16bf kb01 = load_b(&kTile[buf][col][32], g);
    v16bf kb10 = load_b(&kTile[buf][16 + col][0], g);
    v16bf kb11 = load_b(&kTile[buf][16 + col][32], g);
    v8f s0, s1;
#pragma unroll
    for (int r = 0; r < 8; ++r) { s0[r] = 0.f; s1[r] = 0.f; }
    s0 = wmma_bf16(qa0, kb00, s0);
    s0 = wmma_bf16(qa1, kb01, s0);
    s1 = wmma_bf16(qa0, kb10, s1);
    s1 = wmma_bf16(qa1, kb11, s1);

    // V fragments: issue early, consumed after softmax
    v16bf vb0 = load_b(&vTile[buf][col][0], g);
    v16bf vb1 = load_b(&vTile[buf][16 + col][0], g);
    v16bf vb2 = load_b(&vTile[buf][32 + col][0], g);
    v16bf vb3 = load_b(&vTile[buf][48 + col][0], g);

    // softmax in exp2 domain, scale folded
    float mx[8];
#pragma unroll
    for (int r = 0; r < 8; ++r) {
      s0[r] *= SCALE_LOG2E; s1[r] *= SCALE_LOG2E;
      mx[r] = fmaxf(s0[r], s1[r]);
    }
#pragma unroll
    for (int mask = 1; mask <= 8; mask <<= 1)
#pragma unroll
      for (int r = 0; r < 8; ++r) mx[r] = fmaxf(mx[r], __shfl_xor(mx[r], mask, 32));

    float p0[8], p1[8], sum[8];
#pragma unroll
    for (int r = 0; r < 8; ++r) {
      const float mn = fmaxf(m_i[r], mx[r]);
      const float f = fast_exp2(m_i[r] - mn);
      m_i[r] = mn;
      p0[r] = fast_exp2(s0[r] - mn);
      p1[r] = fast_exp2(s1[r] - mn);
      sum[r] = p0[r] + p1[r];
      l_i[r] *= f;
#pragma unroll
      for (int t = 0; t < 4; ++t) o[t][r] *= f;
    }
#pragma unroll
    for (int mask = 1; mask <= 8; mask <<= 1)
#pragma unroll
      for (int r = 0; r < 8; ++r) sum[r] += __shfl_xor(sum[r], mask, 32);
#pragma unroll
    for (int r = 0; r < 8; ++r) l_i[r] += sum[r];

    // D-layout (row=r+8g, col=lane&15) -> LDS -> A-fragment layout
#pragma unroll
    for (int r = 0; r < 8; ++r) {
      pLDS[wave][r + 8 * g][col] = f2bf(p0[r]);
      pLDS[wave][r + 8 * g][16 + col] = f2bf(p1[r]);
    }
    asm volatile("" ::: "memory");
    asm volatile("s_wait_dscnt 0x0" ::: "memory");
    const v16bf pa = load_a(&pLDS[wave][col][0], g);

    o[0] = wmma_bf16(pa, vb0, o[0]);
    o[1] = wmma_bf16(pa, vb1, o[1]);
    o[2] = wmma_bf16(pa, vb2, o[2]);
    o[3] = wmma_bf16(pa, vb3, o[3]);

    wait_async();
    __syncthreads();
  }

#pragma unroll
  for (int r = 0; r < 8; ++r) {
    const float inv = 1.f / l_i[r];
    const int row = q0 + r + 8 * g;
#pragma unroll
    for (int t = 0; t < 4; ++t)
      attn[((size_t)bb * SEQ + row) * DIMC + h * HDIM + t * 16 + col] = f2bf(o[t][r] * inv);
  }
}

// ---------------- projection GEMM: [B*L,DIM] x [DIM,DIM] + bias -> f32 out ----------------
__global__ void __launch_bounds__(256) proj_gemm_kernel(
    const unsigned short* __restrict__ abf,   // [B*L][DIM]
    const unsigned short* __restrict__ wT,    // [DIM][DIM]
    const float* __restrict__ bias,           // [DIM]
    float* __restrict__ out) {                // [B*L][DIM]
  __shared__ unsigned short bTile[2][64][32];
  const int wave = threadIdx.x >> 5;
  const int lane = threadIdx.x & 31;
  const int g = lane >> 4, col = lane & 15;
  const int ngroups = DIMC / 64;                    // 16
  const int nblk = blockIdx.x % ngroups;
  const int mblk = blockIdx.x / ngroups;
  const int m0 = mblk * 128 + wave * 16;
  const int n0 = nblk * 64;

  const int crow = threadIdx.x >> 2;
  const int ccol = (threadIdx.x & 3) * 8;
  const unsigned short* gB = wT + (size_t)(n0 + crow) * DIMC + ccol;

  async_copy_b128(&bTile[0][crow][ccol], gB);
  wait_async();
  __syncthreads();

  v8f acc[4];
#pragma unroll
  for (int t = 0; t < 4; ++t)
#pragma unroll
    for (int r = 0; r < 8; ++r) acc[t][r] = 0.f;

  const unsigned short* arow = abf + (size_t)(m0 + col) * DIMC;
  for (int kt = 0; kt < DIMC; kt += 32) {
    const int buf = (kt >> 5) & 1;
    if (kt + 32 < DIMC)
      async_copy_b128(&bTile[buf ^ 1][crow][ccol], gB + kt + 32);
    v16bf a = load_a(arow + kt, g);
    v16bf b0 = load_b(&bTile[buf][col][0], g);
    v16bf b1 = load_b(&bTile[buf][16 + col][0], g);
    v16bf b2 = load_b(&bTile[buf][32 + col][0], g);
    v16bf b3 = load_b(&bTile[buf][48 + col][0], g);
    acc[0] = wmma_bf16(a, b0, acc[0]);
    acc[1] = wmma_bf16(a, b1, acc[1]);
    acc[2] = wmma_bf16(a, b2, acc[2]);
    acc[3] = wmma_bf16(a, b3, acc[3]);
    wait_async();
    __syncthreads();
  }

#pragma unroll
  for (int t = 0; t < 4; ++t) {
    const float bv = bias[n0 + t * 16 + col];
#pragma unroll
    for (int r = 0; r < 8; ++r)
      out[(size_t)(m0 + r + 8 * g) * DIMC + n0 + t * 16 + col] = acc[t][r] + bv;
  }
}

extern "C" void kernel_launch(void* const* d_in, const int* in_sizes, int n_in,
                              void* d_out, int out_size, void* d_ws, size_t ws_size,
                              hipStream_t stream) {
  const float* x      = (const float*)d_in[0];
  const float* w_qkv  = (const float*)d_in[1];
  const float* b_qkv  = (const float*)d_in[2];
  const float* w_proj = (const float*)d_in[3];
  const float* b_proj = (const float*)d_in[4];
  float* out = (float*)d_out;

  char* ws = (char*)d_ws;
  size_t off = 0;
  auto alloc = [&](size_t bytes) -> void* {
    void* p = ws + off;
    off += (bytes + 255) & ~(size_t)255;
    return p;
  };
  const size_t nx = (size_t)BATCH * SEQ * DIMC;
  unsigned short* xbf    = (unsigned short*)alloc(nx * 2);
  unsigned short* wqkvT  = (unsigned short*)alloc((size_t)3 * DIMC * DIMC * 2);
  unsigned short* wprojT = (unsigned short*)alloc((size_t)DIMC * DIMC * 2);
  unsigned short* Q      = (unsigned short*)alloc(nx * 2);
  unsigned short* K      = (unsigned short*)alloc(nx * 2);
  unsigned short* Vt     = (unsigned short*)alloc(nx * 2);
  unsigned short* attn   = (unsigned short*)alloc(nx * 2);
  (void)ws_size; (void)in_sizes; (void)n_in; (void)out_size;

  {
    int n = (int)nx;
    convert_bf16_kernel<<<(n + 255) / 256, 256, 0, stream>>>(x, xbf, n);
  }
  {
    int n = 3 * DIMC * DIMC;
    transpose_bf16_kernel<<<(n + 255) / 256, 256, 0, stream>>>(w_qkv, wqkvT, DIMC, 3 * DIMC);
  }
  {
    int n = DIMC * DIMC;
    transpose_bf16_kernel<<<(n + 255) / 256, 256, 0, stream>>>(w_proj, wprojT, DIMC, DIMC);
  }
  {
    // (B*L/128) m-blocks x (3*DIM/64) n-blocks = 32*48
    qkv_gemm_kernel<<<32 * 48, 256, 0, stream>>>(xbf, wqkvT, b_qkv, Q, K, Vt);
  }
  {
    // B*H*(L/64) blocks, 4 waves each
    attn_kernel<<<BATCH * NHEAD * (SEQ / 64), 128, 0, stream>>>(Q, K, Vt, attn);
  }
  {
    // (B*L/128) x (DIM/64) = 32*16
    proj_gemm_kernel<<<32 * 16, 256, 0, stream>>>(attn, wprojT, b_proj, out);
  }
}
